// LSTMWithAttention_14422500180690
// MI455X (gfx1250) — compile-verified, matrix operand order checked by reading, not run on hardware
//
#include <hip/hip_runtime.h>
#include <hip/hip_bf16.h>

#define HDIM 512
#define EDIM 256
#define BDIM 64
#define SLEN 128
#define TLEN 48
#define VOUT 32000

typedef __attribute__((ext_vector_type(16))) _Float16 v16h;
typedef __attribute__((ext_vector_type(8)))  _Float16 v8h;
typedef __attribute__((ext_vector_type(8)))  float    v8f;

// ---------------------------------------------------------------------------
// Fragment loader: 16x32 f16 tile of a row-major matrix (row = M for A, N for B)
// Per CDNA5 ISA 7.12.2: lanes 0-15 hold rows 0-15 with K {0..7, 16..23},
// lanes 16-31 hold rows 0-15 with K {8..15, 24..31}.
// ---------------------------------------------------------------------------
__device__ __forceinline__ v16h load_frag16x32(const _Float16* base, int ld,
                                               int row, int k0, int lane) {
  const int hs = (lane >> 4) & 1;
  const _Float16* p = base + (size_t)row * ld + k0 + hs * 8;
  union { v16h v; v8h h8[2]; } u;
  u.h8[0] = *(const v8h*)(p);        // K block 0 half
  u.h8[1] = *(const v8h*)(p + 16);   // K block 1 half
  return u.v;
}

// ---------------------------------------------------------------------------
// Register-blocked WMMA GEMM: each wave computes a 32x64 tile of
// C[M,N] (f32, row-major) = A[M,K] @ B[N,K]^T (+bias / +=C).
// 2 A-frags x 4 B-frags -> 8 v_wmma per K step (one loadcnt wait per 8 MMAs).
// __launch_bounds__(256,1): keep accumulators + pipelined frags in VGPRs
// (CDNA5 waves can hold up to 1024 VGPRs; avoid scratch spills at all cost).
// M multiple of 32, N multiple of 64, K multiple of 32.
// ---------------------------------------------------------------------------
__global__ void __launch_bounds__(256, 1)
gemm_f16f32(const _Float16* __restrict__ A, int lda,
            const _Float16* __restrict__ B, int ldb,
            float* __restrict__ C, int ldc,
            const float* __restrict__ bias,
            int M, int N, int K, int accumulate) {
  const int lane = threadIdx.x & 31;
  const int wave = threadIdx.x >> 5;
  const int ntN  = N >> 6;
  const int numTiles = (M >> 5) * ntN;
  const int tile = blockIdx.x * (blockDim.x >> 5) + wave;
  if (tile >= numTiles) return;
  const int m0 = (tile / ntN) << 5;
  const int n0 = (tile % ntN) << 6;
  const int r  = lane & 15;
  const int hs = lane >> 4;

  v8f acc[2][4];
  if (accumulate) {
#pragma unroll
    for (int i = 0; i < 2; ++i)
#pragma unroll
      for (int j = 0; j < 4; ++j)
#pragma unroll
        for (int g = 0; g < 8; ++g)
          acc[i][j][g] = C[(size_t)(m0 + 16 * i + g + 8 * hs) * ldc + n0 + 16 * j + r];
  } else if (bias) {
#pragma unroll
    for (int j = 0; j < 4; ++j) {
      const float bv = bias[n0 + 16 * j + r];
#pragma unroll
      for (int i = 0; i < 2; ++i)
#pragma unroll
        for (int g = 0; g < 8; ++g) acc[i][j][g] = bv;
    }
  } else {
#pragma unroll
    for (int i = 0; i < 2; ++i)
#pragma unroll
      for (int j = 0; j < 4; ++j)
#pragma unroll
        for (int g = 0; g < 8; ++g) acc[i][j][g] = 0.f;
  }

  for (int k0 = 0; k0 < K; k0 += 32) {
    // Speculative prefetch of next K-slab (silently dropped past the end).
    __builtin_prefetch(A + (size_t)(m0 + r) * lda + k0 + 32, 0, 3);
    __builtin_prefetch(B + (size_t)(n0 + r) * ldb + k0 + 32, 0, 3);
    v16h a0 = load_frag16x32(A, lda, m0 + r,      k0, lane);
    v16h a1 = load_frag16x32(A, lda, m0 + 16 + r, k0, lane);
    v16h b0 = load_frag16x32(B, ldb, n0 + r,      k0, lane);
    v16h b1 = load_frag16x32(B, ldb, n0 + 16 + r, k0, lane);
    v16h b2 = load_frag16x32(B, ldb, n0 + 32 + r, k0, lane);
    v16h b3 = load_frag16x32(B, ldb, n0 + 48 + r, k0, lane);
    acc[0][0] = __builtin_amdgcn_wmma_f32_16x16x32_f16(false, a0, false, b0, (short)0, acc[0][0], false, false);
    acc[0][1] = __builtin_amdgcn_wmma_f32_16x16x32_f16(false, a0, false, b1, (short)0, acc[0][1], false, false);
    acc[0][2] = __builtin_amdgcn_wmma_f32_16x16x32_f16(false, a0, false, b2, (short)0, acc[0][2], false, false);
    acc[0][3] = __builtin_amdgcn_wmma_f32_16x16x32_f16(false, a0, false, b3, (short)0, acc[0][3], false, false);
    acc[1][0] = __builtin_amdgcn_wmma_f32_16x16x32_f16(false, a1, false, b0, (short)0, acc[1][0], false, false);
    acc[1][1] = __builtin_amdgcn_wmma_f32_16x16x32_f16(false, a1, false, b1, (short)0, acc[1][1], false, false);
    acc[1][2] = __builtin_amdgcn_wmma_f32_16x16x32_f16(false, a1, false, b2, (short)0, acc[1][2], false, false);
    acc[1][3] = __builtin_amdgcn_wmma_f32_16x16x32_f16(false, a1, false, b3, (short)0, acc[1][3], false, false);
  }

#pragma unroll
  for (int i = 0; i < 2; ++i)
#pragma unroll
    for (int j = 0; j < 4; ++j)
#pragma unroll
      for (int g = 0; g < 8; ++g)
        C[(size_t)(m0 + 16 * i + g + 8 * hs) * ldc + n0 + 16 * j + r] = acc[i][j][g];
}

// ---------------------------------------------------------------------------
// Output projection GEMM (32x64 wave tiles) with permuted store:
// A = dec_hist [T*B, 3H] f16 ; B = out_W [VOUT, 3H] f16 ; out[b,t,n] f32
// row m = t*B + b  ->  out[b*T*VOUT + t*VOUT + n]
// ---------------------------------------------------------------------------
__global__ void __launch_bounds__(256, 1)
gemm_out_proj(const _Float16* __restrict__ A,
              const _Float16* __restrict__ B,
              const float* __restrict__ bias,
              float* __restrict__ out) {
  const int lane = threadIdx.x & 31;
  const int wave = threadIdx.x >> 5;
  const int ntN  = VOUT >> 6;                      // 500
  const int numTiles = ((TLEN * BDIM) >> 5) * ntN; // 96 * 500
  const int tile = blockIdx.x * (blockDim.x >> 5) + wave;
  if (tile >= numTiles) return;
  const int m0 = (tile / ntN) << 5;
  const int n0 = (tile % ntN) << 6;
  const int r  = lane & 15;
  const int hs = lane >> 4;
  const int K  = 3 * HDIM, lda = 3 * HDIM, ldb = 3 * HDIM;

  v8f acc[2][4];
#pragma unroll
  for (int j = 0; j < 4; ++j) {
    const float bv = bias[n0 + 16 * j + r];
#pragma unroll
    for (int i = 0; i < 2; ++i)
#pragma unroll
      for (int g = 0; g < 8; ++g) acc[i][j][g] = bv;
  }

  for (int k0 = 0; k0 < K; k0 += 32) {
    __builtin_prefetch(A + (size_t)(m0 + r) * lda + k0 + 32, 0, 3);
    __builtin_prefetch(B + (size_t)(n0 + r) * ldb + k0 + 32, 0, 3);
    v16h a0 = load_frag16x32(A, lda, m0 + r,      k0, lane);
    v16h a1 = load_frag16x32(A, lda, m0 + 16 + r, k0, lane);
    v16h b0 = load_frag16x32(B, ldb, n0 + r,      k0, lane);
    v16h b1 = load_frag16x32(B, ldb, n0 + 16 + r, k0, lane);
    v16h b2 = load_frag16x32(B, ldb, n0 + 32 + r, k0, lane);
    v16h b3 = load_frag16x32(B, ldb, n0 + 48 + r, k0, lane);
    acc[0][0] = __builtin_amdgcn_wmma_f32_16x16x32_f16(false, a0, false, b0, (short)0, acc[0][0], false, false);
    acc[0][1] = __builtin_amdgcn_wmma_f32_16x16x32_f16(false, a0, false, b1, (short)0, acc[0][1], false, false);
    acc[0][2] = __builtin_amdgcn_wmma_f32_16x16x32_f16(false, a0, false, b2, (short)0, acc[0][2], false, false);
    acc[0][3] = __builtin_amdgcn_wmma_f32_16x16x32_f16(false, a0, false, b3, (short)0, acc[0][3], false, false);
    acc[1][0] = __builtin_amdgcn_wmma_f32_16x16x32_f16(false, a1, false, b0, (short)0, acc[1][0], false, false);
    acc[1][1] = __builtin_amdgcn_wmma_f32_16x16x32_f16(false, a1, false, b1, (short)0, acc[1][1], false, false);
    acc[1][2] = __builtin_amdgcn_wmma_f32_16x16x32_f16(false, a1, false, b2, (short)0, acc[1][2], false, false);
    acc[1][3] = __builtin_amdgcn_wmma_f32_16x16x32_f16(false, a1, false, b3, (short)0, acc[1][3], false, false);
  }

#pragma unroll
  for (int i = 0; i < 2; ++i)
#pragma unroll
    for (int g = 0; g < 8; ++g) {
      const int m = m0 + 16 * i + g + 8 * hs;
      const int t = m >> 6;       // / BDIM
      const int b = m & 63;       // % BDIM
      float* orow = out + (size_t)b * (TLEN * VOUT) + (size_t)t * VOUT;
#pragma unroll
      for (int j = 0; j < 4; ++j) orow[n0 + 16 * j + r] = acc[i][j][g];
    }
}

// ---------------------------------------------------------------------------
// Elementwise helpers
// ---------------------------------------------------------------------------
__global__ void f32_to_f16(const float* __restrict__ s, _Float16* __restrict__ d, int n) {
  int i = blockIdx.x * blockDim.x + threadIdx.x;
  if (i < n) d[i] = (_Float16)s[i];
}
__global__ void zero_f32(float* __restrict__ p, int n) {
  int i = blockIdx.x * blockDim.x + threadIdx.x;
  if (i < n) p[i] = 0.f;
}
__global__ void zero_f16(_Float16* __restrict__ p, int n) {
  int i = blockIdx.x * blockDim.x + threadIdx.x;
  if (i < n) p[i] = (_Float16)0.f;
}

// Gather embeddings, time-major: out[(s*B + b)*E + e] = emb[tok[b,s], e]
__global__ void gather_embed(const int* __restrict__ toks, const float* __restrict__ emb,
                             _Float16* __restrict__ out, int L) {
  int idx = blockIdx.x * blockDim.x + threadIdx.x;
  int row = idx >> 8;                     // EDIM = 256
  if (row >= L * BDIM) return;
  int e = idx & (EDIM - 1);
  int s = row / BDIM, b = row % BDIM;
  int tok = toks[b * L + s];
  out[idx] = (_Float16)emb[(size_t)tok * EDIM + e];
}

// LSTM cell pointwise: gates = xW + hW + bias (torch order i,f,g,o)
__global__ void lstm_cell(const float* __restrict__ xW, const float* __restrict__ hW,
                          const float* __restrict__ bias, float* __restrict__ c,
                          _Float16* __restrict__ h16,
                          float* __restrict__ dst32, int stride32,
                          _Float16* __restrict__ dst16, int stride16) {
  int idx = blockIdx.x * blockDim.x + threadIdx.x;
  if (idx >= BDIM * HDIM) return;
  int b = idx >> 9, j = idx & (HDIM - 1);
  const float* gx = xW + (size_t)b * 4 * HDIM;
  const float* gh = hW + (size_t)b * 4 * HDIM;
  float gi = gx[j]            + gh[j]            + bias[j];
  float gf = gx[HDIM + j]     + gh[HDIM + j]     + bias[HDIM + j];
  float gg = gx[2 * HDIM + j] + gh[2 * HDIM + j] + bias[2 * HDIM + j];
  float go = gx[3 * HDIM + j] + gh[3 * HDIM + j] + bias[3 * HDIM + j];
  float si = 1.f / (1.f + expf(-gi));
  float sf = 1.f / (1.f + expf(-gf));
  float so = 1.f / (1.f + expf(-go));
  float cv = sf * c[idx] + si * tanhf(gg);
  c[idx] = cv;
  float hv = so * tanhf(cv);
  h16[idx] = (_Float16)hv;
  if (dst32) dst32[(size_t)b * stride32 + j] = hv;
  if (dst16) dst16[(size_t)b * stride16 + j] = (_Float16)hv;
}

// Bahdanau attention for one decoder step, one block per batch row (128 threads):
// score_s = v . tanh(encPart[b,s,:] + hW[b,:]); softmax over s; ctx = sum_s w_s * encOut[b,s,:]
__global__ void attention_step(const float* __restrict__ hW, const float* __restrict__ encPart,
                               const float* __restrict__ v, const float* __restrict__ encOut,
                               float* __restrict__ ctx32, _Float16* __restrict__ ctx16,
                               _Float16* __restrict__ histCtx) {
  const int b = blockIdx.x, tid = threadIdx.x;    // blockDim = 128 = SLEN
  __shared__ float sh[HDIM];
  __shared__ float sv[HDIM];
  __shared__ float sw[SLEN];
  __shared__ float red[SLEN];
  for (int j = tid; j < HDIM; j += SLEN) { sh[j] = hW[b * HDIM + j]; sv[j] = v[j]; }
  __syncthreads();

  const float* ep = encPart + ((size_t)b * SLEN + tid) * HDIM;
  float e = 0.f;
  for (int j = 0; j < HDIM; ++j) e += tanhf(ep[j] + sh[j]) * sv[j];

  red[tid] = e; __syncthreads();
  for (int o = SLEN / 2; o > 0; o >>= 1) {
    if (tid < o) red[tid] = fmaxf(red[tid], red[tid + o]);
    __syncthreads();
  }
  float mx = red[0]; __syncthreads();
  float ex = expf(e - mx);
  red[tid] = ex; __syncthreads();
  for (int o = SLEN / 2; o > 0; o >>= 1) {
    if (tid < o) red[tid] += red[tid + o];
    __syncthreads();
  }
  float inv = 1.f / red[0]; __syncthreads();
  sw[tid] = ex * inv;
  __syncthreads();

  for (int d = tid; d < 2 * HDIM; d += SLEN) {
    const float* eo = encOut + (size_t)b * SLEN * 2 * HDIM + d;
    float a = 0.f;
    for (int s = 0; s < SLEN; ++s) a += sw[s] * eo[(size_t)s * 2 * HDIM];
    ctx32[b * 2 * HDIM + d] = a;
    _Float16 hv = (_Float16)a;
    ctx16[b * 2 * HDIM + d] = hv;
    histCtx[(size_t)b * (3 * HDIM) + d] = hv;
  }
}

// ---------------------------------------------------------------------------
// Host orchestration
// ---------------------------------------------------------------------------
static inline size_t alignUp(size_t x) { return (x + 255) & ~(size_t)255; }

extern "C" void kernel_launch(void* const* d_in, const int* in_sizes, int n_in,
                              void* d_out, int out_size, void* d_ws, size_t ws_size,
                              hipStream_t stream) {
  (void)in_sizes; (void)n_in; (void)out_size; (void)ws_size;
  const int*   jpTok   = (const int*)d_in[0];
  const int*   enTok   = (const int*)d_in[1];
  const float* embJP   = (const float*)d_in[2];
  const float* embEN   = (const float*)d_in[3];
  const float* eWihF   = (const float*)d_in[4];
  const float* eWhhF   = (const float*)d_in[5];
  const float* eBF     = (const float*)d_in[6];
  const float* eWihB   = (const float*)d_in[7];
  const float* eWhhB   = (const float*)d_in[8];
  const float* eBB     = (const float*)d_in[9];
  const float* dWih    = (const float*)d_in[10];
  const float* dWhh    = (const float*)d_in[11];
  const float* dB      = (const float*)d_in[12];
  const float* attnW   = (const float*)d_in[13];
  const float* attnV   = (const float*)d_in[14];
  const float* outW    = (const float*)d_in[15];
  const float* outB    = (const float*)d_in[16];
  float* out = (float*)d_out;

  // ---- carve workspace ----
  char* w = (char*)d_ws; size_t off = 0;
  auto takeH = [&](size_t n) { _Float16* p = (_Float16*)(w + off); off = alignUp(off + n * 2); return p; };
  auto takeF = [&](size_t n) { float* p = (float*)(w + off); off = alignUp(off + n * 4); return p; };

  _Float16* wXe     = takeH((size_t)SLEN * BDIM * EDIM);
  _Float16* wYe     = takeH((size_t)TLEN * BDIM * EDIM);
  _Float16* wEWihF  = takeH((size_t)4 * HDIM * EDIM);
  _Float16* wEWhhF  = takeH((size_t)4 * HDIM * HDIM);
  _Float16* wEWihB  = takeH((size_t)4 * HDIM * EDIM);
  _Float16* wEWhhB  = takeH((size_t)4 * HDIM * HDIM);
  _Float16* wDWih   = takeH((size_t)4 * HDIM * (EDIM + 2 * HDIM));
  _Float16* wDWhh   = takeH((size_t)4 * HDIM * HDIM);
  _Float16* wAttnW  = takeH((size_t)HDIM * 3 * HDIM);
  _Float16* wOutW   = takeH((size_t)VOUT * 3 * HDIM);
  _Float16* wEncO16 = takeH((size_t)BDIM * SLEN * 2 * HDIM);
  _Float16* wHF16   = takeH((size_t)BDIM * HDIM);
  _Float16* wHB16   = takeH((size_t)BDIM * HDIM);          // also decoder h
  _Float16* wCtx16  = takeH((size_t)BDIM * 2 * HDIM);
  _Float16* wHist   = takeH((size_t)TLEN * BDIM * 3 * HDIM);

  float* wXWf   = takeF((size_t)SLEN * BDIM * 4 * HDIM);
  float* wXWb   = takeF((size_t)SLEN * BDIM * 4 * HDIM);
  float* wYW    = takeF((size_t)TLEN * BDIM * 4 * HDIM);
  float* wEncO  = takeF((size_t)BDIM * SLEN * 2 * HDIM);
  float* wEPart = takeF((size_t)BDIM * SLEN * HDIM);
  float* wGates = takeF((size_t)BDIM * 4 * HDIM);
  float* wCF    = takeF((size_t)BDIM * HDIM);
  float* wCB    = takeF((size_t)BDIM * HDIM);              // also decoder c
  float* wHWat  = takeF((size_t)BDIM * HDIM);
  float* wCtx32 = takeF((size_t)BDIM * 2 * HDIM);

  auto cvt = [&](const float* s, _Float16* d, size_t n) {
    f32_to_f16<<<(unsigned)((n + 255) / 256), 256, 0, stream>>>(s, d, (int)n);
  };
  // 32x64 wave tiles, 8 waves (256 threads) per block
  auto gemm = [&](const _Float16* A, int lda, const _Float16* B, int ldb,
                  float* C, int ldc, const float* bias, int M, int N, int K, int acc) {
    int tiles = (M >> 5) * (N >> 6);
    gemm_f16f32<<<(tiles + 7) >> 3, 256, 0, stream>>>(A, lda, B, ldb, C, ldc, bias, M, N, K, acc);
  };

  // ---- init recurrent state ----
  zero_f32<<<(BDIM * HDIM) / 256, 256, 0, stream>>>(wCF, BDIM * HDIM);
  zero_f32<<<(BDIM * HDIM) / 256, 256, 0, stream>>>(wCB, BDIM * HDIM);
  zero_f16<<<(BDIM * HDIM) / 256, 256, 0, stream>>>(wHF16, BDIM * HDIM);
  zero_f16<<<(BDIM * HDIM) / 256, 256, 0, stream>>>(wHB16, BDIM * HDIM);

  // ---- convert weights to f16 ----
  cvt(eWihF, wEWihF, (size_t)4 * HDIM * EDIM);
  cvt(eWhhF, wEWhhF, (size_t)4 * HDIM * HDIM);
  cvt(eWihB, wEWihB, (size_t)4 * HDIM * EDIM);
  cvt(eWhhB, wEWhhB, (size_t)4 * HDIM * HDIM);
  cvt(dWih,  wDWih,  (size_t)4 * HDIM * (EDIM + 2 * HDIM));
  cvt(dWhh,  wDWhh,  (size_t)4 * HDIM * HDIM);
  cvt(attnW, wAttnW, (size_t)HDIM * 3 * HDIM);
  cvt(outW,  wOutW,  (size_t)VOUT * 3 * HDIM);

  // ---- embedding gathers (time-major) ----
  gather_embed<<<(SLEN * BDIM * EDIM) / 256, 256, 0, stream>>>(jpTok, embJP, wXe, SLEN);
  gather_embed<<<(TLEN * BDIM * EDIM) / 256, 256, 0, stream>>>(enTok, embEN, wYe, TLEN);

  // ---- hoisted input-projection GEMMs ----
  gemm(wXe, EDIM, wEWihF, EDIM, wXWf, 4 * HDIM, nullptr, SLEN * BDIM, 4 * HDIM, EDIM, 0);
  gemm(wXe, EDIM, wEWihB, EDIM, wXWb, 4 * HDIM, nullptr, SLEN * BDIM, 4 * HDIM, EDIM, 0);
  // decoder embedding part: dec_Wih[:, :E]
  gemm(wYe, EDIM, wDWih, EDIM + 2 * HDIM, wYW, 4 * HDIM, nullptr, TLEN * BDIM, 4 * HDIM, EDIM, 0);

  // ---- encoder recurrence (fwd + bwd interleaved on the stream) ----
  const int lcBlocks = (BDIM * HDIM) / 256;
  for (int s = 0; s < SLEN; ++s) {
    // forward dir
    gemm(wHF16, HDIM, wEWhhF, HDIM, wGates, 4 * HDIM, nullptr, BDIM, 4 * HDIM, HDIM, 0);
    lstm_cell<<<lcBlocks, 256, 0, stream>>>(
        wXWf + (size_t)s * BDIM * 4 * HDIM, wGates, eBF, wCF, wHF16,
        wEncO + (size_t)s * 2 * HDIM, SLEN * 2 * HDIM,
        wEncO16 + (size_t)s * 2 * HDIM, SLEN * 2 * HDIM);
    // backward dir (processes xe[S-1-s], writes position S-1-s, high half)
    const int sr = SLEN - 1 - s;
    gemm(wHB16, HDIM, wEWhhB, HDIM, wGates, 4 * HDIM, nullptr, BDIM, 4 * HDIM, HDIM, 0);
    lstm_cell<<<lcBlocks, 256, 0, stream>>>(
        wXWb + (size_t)sr * BDIM * 4 * HDIM, wGates, eBB, wCB, wHB16,
        wEncO + (size_t)sr * 2 * HDIM + HDIM, SLEN * 2 * HDIM,
        wEncO16 + (size_t)sr * 2 * HDIM + HDIM, SLEN * 2 * HDIM);
  }

  // ---- attention pre-projection: encPart = enc_out @ W_e^T (step-invariant) ----
  gemm(wEncO16, 2 * HDIM, wAttnW + HDIM, 3 * HDIM, wEPart, HDIM, nullptr,
       BDIM * SLEN, HDIM, 2 * HDIM, 0);

  // ---- decoder recurrence (h,c initialized to backward-final states) ----
  for (int t = 0; t < TLEN; ++t) {
    // h @ W_h^T for attention energies
    gemm(wHB16, HDIM, wAttnW, 3 * HDIM, wHWat, HDIM, nullptr, BDIM, HDIM, HDIM, 0);
    attention_step<<<BDIM, SLEN, 0, stream>>>(
        wHWat, wEPart, attnV, wEncO, wCtx32, wCtx16,
        wHist + (size_t)t * BDIM * 3 * HDIM + HDIM);
    // gates = ctx @ Wih_ctx^T  (+=)  h @ Whh^T
    gemm(wCtx16, 2 * HDIM, wDWih + EDIM, EDIM + 2 * HDIM, wGates, 4 * HDIM, nullptr,
         BDIM, 4 * HDIM, 2 * HDIM, 0);
    gemm(wHB16, HDIM, wDWhh, HDIM, wGates, 4 * HDIM, nullptr, BDIM, 4 * HDIM, HDIM, 1);
    lstm_cell<<<lcBlocks, 256, 0, stream>>>(
        wYW + (size_t)t * BDIM * 4 * HDIM, wGates, dB, wCB, wHB16,
        nullptr, 0, wHist + (size_t)t * BDIM * 3 * HDIM, 3 * HDIM);
  }

  // ---- deferred output projection: [T*B,3H] @ [VOUT,3H]^T + b, permuted store ----
  {
    int tiles = ((TLEN * BDIM) >> 5) * (VOUT >> 6);
    gemm_out_proj<<<(tiles + 7) >> 3, 256, 0, stream>>>(wHist, wOutW, outB, out);
  }
}
